// VRNN_79680233276303
// MI455X (gfx1250) — matrix operand order repaired and checked by reading, not compiled, not run on hardware
//
#include <hip/hip_runtime.h>
#include <math.h>

// ---------------------------------------------------------------------------
// VRNN forward for MI455X (gfx1250, wave32).
// All conv / GRU matmuls are lowered to v_wmma_f32_16x16x32_f16, with each
// wave computing a 16(M) x 64(N) output tile: one A fragment is reused by
// four B fragments. The K-loop is manually software-pipelined (double-
// buffered fragments) so next-iteration global_load_b128s overlap the four
// in-flight WMMAs instead of serializing on s_wait_loadcnt 0.
// A-matrices (weights) are converted once per launch to an f16 layout
// kept row-major (K unpermuted, padded to Kp) so each lane issues one
// contiguous 32B load per K-step. B-matrices (im2col panels / packed
// activations) are stored column-major [n][Kp] so lane n's 16 K-halves are
// contiguous as well. Transposed-conv kernels stage their small weight
// tensors in LDS (ds path) before the inner loops.
// ---------------------------------------------------------------------------

typedef __attribute__((ext_vector_type(16))) _Float16 v16h;
typedef __attribute__((ext_vector_type(8)))  float    v8f;

#define ACT_NONE     0
#define ACT_RELU     1
#define ACT_SOFTPLUS 2

#define GEMM_WAVES 8
#define BATCH 128

__device__ __forceinline__ float apply_act(float v, int act) {
  if (act == ACT_RELU) return fmaxf(v, 0.0f);
  if (act == ACT_SOFTPLUS) return logf(1.0f + expf(v));
  return v;
}

// --------------------------- WMMA GEMM -------------------------------------
// D[M,N] = act(A[M,K] * B[K,N] + bias[M]), one 16x64 tile per wave.
// A: f16, rows of Kp halves, K-order plain (not swizzled), zero-padded from K to Kp
//    -> lanes 0-15 read halves [k0+0,16), lanes 16-31 read [k0+16,32) contiguously.
// B: f16 column-major Bc[n*Kp + k] (this kernel's per-lane K order is contiguous, k = 16*hi + s, same as A).
// layout==0: out[m*N+n] row-major.  layout==1: NCHW scatter, n = b*S + s,
//            out[(b*M + m)*S + s]  (M == Cout).  Requires N % 64 == 0.
__global__ void __launch_bounds__(GEMM_WAVES * 32, 1)
wmma_gemm_k(const _Float16* __restrict__ A,
            const _Float16* __restrict__ Bc,
            const float* __restrict__ bias,
            float* __restrict__ out,
            int M, int N, int Kp, int act, int layout, int S) {
  const int lane = threadIdx.x & 31;
  const int wave = threadIdx.x >> 5;
  const int tilesN4 = N >> 6;                     // groups of four 16-wide n-tiles
  const int tilesM  = M >> 4;
  const int tile = blockIdx.x * GEMM_WAVES + wave;
  if (tile >= tilesM * tilesN4) return;           // wave-uniform: EXEC all-1s for WMMA
  const int tm = tile / tilesN4;
  const int tg = tile - tm * tilesN4;
  const int hi = lane >> 4;                       // 0: lanes 0-15, 1: lanes 16-31
  const int l  = lane & 15;

  const _Float16* aptr = A  + (size_t)(tm * 16 + l) * Kp + (hi << 4);
  const _Float16* bptr = Bc + (size_t)(tg * 64 + l) * Kp + (hi << 4);
  const size_t bstep = (size_t)16 * Kp;           // next 16-column subtile

  // prologue: fetch first K-slice fragments
  v16h av  = *(const v16h*)(aptr);
  v16h bv0 = *(const v16h*)(bptr);
  v16h bv1 = *(const v16h*)(bptr + bstep);
  v16h bv2 = *(const v16h*)(bptr + 2 * bstep);
  v16h bv3 = *(const v16h*)(bptr + 3 * bstep);

  v8f acc0 = {}, acc1 = {}, acc2 = {}, acc3 = {};
  const int nk = Kp >> 5;
  for (int i = 1; i < nk; ++i) {
    const int k0 = i << 5;
    __builtin_prefetch(aptr + k0 + 224, 0, 1);    // global_prefetch_b8
    // issue next-iteration loads before consuming current fragments
    v16h av_n  = *(const v16h*)(aptr + k0);
    v16h bv0_n = *(const v16h*)(bptr + k0);
    v16h bv1_n = *(const v16h*)(bptr + bstep + k0);
    v16h bv2_n = *(const v16h*)(bptr + 2 * bstep + k0);
    v16h bv3_n = *(const v16h*)(bptr + 3 * bstep + k0);
    acc0 = __builtin_amdgcn_wmma_f32_16x16x32_f16(false, av, false, bv0, (short)0, acc0, false, false);
    acc1 = __builtin_amdgcn_wmma_f32_16x16x32_f16(false, av, false, bv1, (short)0, acc1, false, false);
    acc2 = __builtin_amdgcn_wmma_f32_16x16x32_f16(false, av, false, bv2, (short)0, acc2, false, false);
    acc3 = __builtin_amdgcn_wmma_f32_16x16x32_f16(false, av, false, bv3, (short)0, acc3, false, false);
    av = av_n; bv0 = bv0_n; bv1 = bv1_n; bv2 = bv2_n; bv3 = bv3_n;
  }
  acc0 = __builtin_amdgcn_wmma_f32_16x16x32_f16(false, av, false, bv0, (short)0, acc0, false, false);
  acc1 = __builtin_amdgcn_wmma_f32_16x16x32_f16(false, av, false, bv1, (short)0, acc1, false, false);
  acc2 = __builtin_amdgcn_wmma_f32_16x16x32_f16(false, av, false, bv2, (short)0, acc2, false, false);
  acc3 = __builtin_amdgcn_wmma_f32_16x16x32_f16(false, av, false, bv3, (short)0, acc3, false, false);

  const int m0 = tm * 16 + (hi << 3);             // C/D: VGPR r -> M = m0 + r
#pragma unroll
  for (int j = 0; j < 4; ++j) {
    const v8f acc = (j == 0) ? acc0 : (j == 1) ? acc1 : (j == 2) ? acc2 : acc3;
    const int n = tg * 64 + j * 16 + l;
#pragma unroll
    for (int r = 0; r < 8; ++r) {
      const int m = m0 + r;
      float v = apply_act(acc[r] + bias[m], act);
      if (layout == 0) {
        out[(size_t)m * N + n] = v;
      } else {
        const int b = n / S;
        const int s = n - b * S;
        out[((size_t)b * M + m) * S + s] = v;
      }
    }
  }
}

// ------------------- weight f32 -> f16 padded A panel ---------------------
// dst[m][pos] = (f16)src[m][pos] for pos < K, else 0; K order is not permuted.
__global__ void wswz_k(const float* __restrict__ src, int M, int K, int Kp,
                       _Float16* __restrict__ dst) {
  long gid = (long)blockIdx.x * blockDim.x + threadIdx.x;
  long total = (long)M * Kp;
  if (gid >= total) return;
  int pos = (int)(gid % Kp);
  int m   = (int)(gid / Kp);
  int kb  = pos >> 5;
  int p   = pos & 31;
  int g   = p >> 3;
  int korig = kb * 32 + p;
  dst[gid] = (korig < K) ? (_Float16)src[(long)m * K + korig] : (_Float16)0.0f;
}

// --------------------------- im2col (f16, col-major) ------------------------
// Supports channel-concat of two sources (s1: C1 ch, s2: C2 ch) and a batch
// stride/offset on s1 so frames of x[B,T,3,32,32] can be read in place.
__global__ void im2col_k(const float* __restrict__ s1, int C1, long s1_bstride, long s1_off,
                         const float* __restrict__ s2, int C2,
                         int H, int W, int OH, int OW, int stride, int pad,
                         int B, int Kp, _Float16* __restrict__ Bc) {
  long gid = (long)blockIdx.x * blockDim.x + threadIdx.x;
  long total = (long)B * OH * OW * Kp;
  if (gid >= total) return;
  int  k = (int)(gid % Kp);
  long n = gid / Kp;
  int ox = (int)(n % OW);
  int oy = (int)((n / OW) % OH);
  int b  = (int)(n / ((long)OW * OH));
  float val = 0.0f;
  int c = k / 9;
  if (c < C1 + C2) {
    int rem = k - c * 9;
    int ky = rem / 3, kx = rem - ky * 3;
    int iy = oy * stride - pad + ky;
    int ix = ox * stride - pad + kx;
    if (iy >= 0 && iy < H && ix >= 0 && ix < W) {
      if (c < C1) val = s1[s1_off + (long)b * s1_bstride + ((long)c * H + iy) * W + ix];
      else        val = s2[(((long)b * C2 + (c - C1)) * H + iy) * W + ix];
    }
  }
  Bc[(long)n * Kp + k] = (_Float16)val;
}

// --------------------- GRU input packing (col-major f16) --------------------
__global__ void pack_gi_k(const float* __restrict__ px2, const float* __restrict__ zpsi,
                          const float* __restrict__ h, _Float16* __restrict__ Bc, int B) {
  long gid = (long)blockIdx.x * blockDim.x + threadIdx.x;
  if (gid >= (long)B * 3072) return;
  int k = (int)(gid % 3072);
  int b = (int)(gid / 3072);
  int c = k >> 6, s = k & 63;
  float v;
  if      (c < 16) v = px2 [((long)b * 16 + c)        * 64 + s];
  else if (c < 32) v = zpsi[((long)b * 16 + (c - 16)) * 64 + s];
  else             v = h   [(long)b * 1024 + (c - 32) * 64 + s];
  Bc[gid] = (_Float16)v;
}

__global__ void f32tof16_k(const float* __restrict__ src, _Float16* __restrict__ dst, long n) {
  long gid = (long)blockIdx.x * blockDim.x + threadIdx.x;
  if (gid < n) dst[gid] = (_Float16)src[gid];
}

__global__ void zero_k(float* __restrict__ p, long n) {
  long gid = (long)blockIdx.x * blockDim.x + threadIdx.x;
  if (gid < n) p[gid] = 0.0f;
}

// z = zm + zlv * eps  (source's literal reparam)
__global__ void zcomb_k(const float* __restrict__ zm, const float* __restrict__ zlv,
                        const float* __restrict__ eps, float* __restrict__ z, long n) {
  long gid = (long)blockIdx.x * blockDim.x + threadIdx.x;
  if (gid < n) z[gid] = fmaf(zlv[gid], eps[gid], zm[gid]);
}

// ------------------------- transposed convolutions --------------------------
// dec1: in = concat(h[16], zpsi[16]) 8x8, w[in=32][out=32][3][3], s=2, p=1 -> 15x15, ReLU
// Weights (32*32*9 = 9216 f32) staged in LDS once per block.
__global__ void convt1_k(const float* __restrict__ h, const float* __restrict__ zpsi,
                         const float* __restrict__ w, const float* __restrict__ bias,
                         float* __restrict__ out, int B) {
  __shared__ float sw[32 * 32 * 9];
  for (int i = threadIdx.x; i < 32 * 32 * 9; i += blockDim.x) sw[i] = w[i];
  __syncthreads();

  long gid = (long)blockIdx.x * blockDim.x + threadIdx.x;
  if (gid >= (long)B * 32 * 225) return;
  int s = (int)(gid % 225);
  int o = (int)((gid / 225) % 32);
  int b = (int)(gid / (225L * 32));
  int oy = s / 15, ox = s - oy * 15;
  float acc = bias[o];
  for (int ky = 0; ky < 3; ++ky) {
    int ty = oy + 1 - ky;
    if (ty < 0 || (ty & 1)) continue;
    int iy = ty >> 1; if (iy >= 8) continue;
    for (int kx = 0; kx < 3; ++kx) {
      int tx = ox + 1 - kx;
      if (tx < 0 || (tx & 1)) continue;
      int ix = tx >> 1; if (ix >= 8) continue;
      for (int i = 0; i < 32; ++i) {
        float inv = (i < 16) ? h   [((long)b * 16 + i)        * 64 + iy * 8 + ix]
                             : zpsi[((long)b * 16 + (i - 16)) * 64 + iy * 8 + ix];
        acc = fmaf(sw[((i * 32) + o) * 9 + ky * 3 + kx], inv, acc);
      }
    }
  }
  out[gid] = fmaxf(acc, 0.0f);
}

// dec2: in [B,32,15,15], w[32][3][3][3], s=2, p=0, out_pad=1 -> [B,3,32,32], sigmoid.
// Writes both the recurrent dec buffer and the final output slice for step t.
// Weights (32*3*9 = 864 f32) staged in LDS once per block.
__global__ void convt2_k(const float* __restrict__ in, const float* __restrict__ w,
                         const float* __restrict__ bias, float* __restrict__ dec,
                         float* __restrict__ outbuf, int B, int t) {
  __shared__ float sw[32 * 3 * 9];
  for (int i = threadIdx.x; i < 32 * 3 * 9; i += blockDim.x) sw[i] = w[i];
  __syncthreads();

  long gid = (long)blockIdx.x * blockDim.x + threadIdx.x;
  if (gid >= (long)B * 3 * 1024) return;
  int s = (int)(gid % 1024);
  int o = (int)((gid / 1024) % 3);
  int b = (int)(gid / 3072);
  int oy = s >> 5, ox = s & 31;
  float acc = bias[o];
  for (int ky = 0; ky < 3; ++ky) {
    int ty = oy - ky;
    if (ty < 0 || (ty & 1)) continue;
    int iy = ty >> 1; if (iy >= 15) continue;
    for (int kx = 0; kx < 3; ++kx) {
      int tx = ox - kx;
      if (tx < 0 || (tx & 1)) continue;
      int ix = tx >> 1; if (ix >= 15) continue;
      for (int i = 0; i < 32; ++i)
        acc = fmaf(sw[((i * 3) + o) * 9 + ky * 3 + kx],
                   in[((long)b * 32 + i) * 225 + iy * 15 + ix], acc);
    }
  }
  float v = 1.0f / (1.0f + expf(-acc));
  dec[gid] = v;
  outbuf[(((long)b * 27 + t) * 3 + o) * 1024 + s] = v;
}

// ------------------------------ GRU gates -----------------------------------
// gih/ghh stored [3072, B] row-major (GEMM layout 0).
__global__ void gruc_k(const float* __restrict__ gih, const float* __restrict__ ghh,
                       const float* __restrict__ hprev, float* __restrict__ hnew, int B) {
  long gid = (long)blockIdx.x * blockDim.x + threadIdx.x;
  if (gid >= (long)B * 1024) return;
  int m = (int)(gid % 1024);
  int b = (int)(gid / 1024);
  float ir = gih[(long)m * B + b],          hr = ghh[(long)m * B + b];
  float iz = gih[(long)(m + 1024) * B + b], hz = ghh[(long)(m + 1024) * B + b];
  float in_ = gih[(long)(m + 2048) * B + b], hn = ghh[(long)(m + 2048) * B + b];
  float r = 1.0f / (1.0f + expf(-(ir + hr)));
  float u = 1.0f / (1.0f + expf(-(iz + hz)));
  float ng = tanhf(in_ + r * hn);
  float hp = hprev[(long)b * 1024 + m];
  hnew[(long)b * 1024 + m] = (1.0f - u) * ng + u * hp;
}

// ------------------------------- host side ----------------------------------
extern "C" void kernel_launch(void* const* d_in, const int* in_sizes, int n_in,
                              void* d_out, int out_size, void* d_ws, size_t ws_size,
                              hipStream_t stream) {
  const float* x       = (const float*)d_in[0];
  const float* eps     = (const float*)d_in[1];
  const float* psix1_w = (const float*)d_in[2];
  const float* psix1_b = (const float*)d_in[3];
  const float* psix2_w = (const float*)d_in[4];
  const float* psix2_b = (const float*)d_in[5];
  const float* psiz_w  = (const float*)d_in[6];
  const float* psiz_b  = (const float*)d_in[7];
  const float* enc1_w  = (const float*)d_in[8];
  const float* enc1_b  = (const float*)d_in[9];
  const float* enc2_w  = (const float*)d_in[10];
  const float* enc2_b  = (const float*)d_in[11];
  const float* encm_w  = (const float*)d_in[12];
  const float* encm_b  = (const float*)d_in[13];
  const float* encv_w  = (const float*)d_in[14];
  const float* encv_b  = (const float*)d_in[15];
  const float* pri_w   = (const float*)d_in[16];
  const float* pri_b   = (const float*)d_in[17];
  const float* prim_w  = (const float*)d_in[18];
  const float* prim_b  = (const float*)d_in[19];
  const float* priv_w  = (const float*)d_in[20];
  const float* priv_b  = (const float*)d_in[21];
  const float* dec1_w  = (const float*)d_in[22];
  const float* dec1_b  = (const float*)d_in[23];
  const float* dec2_w  = (const float*)d_in[24];
  const float* dec2_b  = (const float*)d_in[25];
  const float* gru_wi  = (const float*)d_in[26];
  const float* gru_wh  = (const float*)d_in[27];
  const float* gru_bi  = (const float*)d_in[28];
  const float* gru_bh  = (const float*)d_in[29];
  float* out = (float*)d_out;

  const int B = BATCH, T = 28, NSTEP = 27;

  // ---- workspace suballocation (256B-aligned so v16h loads stay aligned) ----
  char* wsp = (char*)d_ws;
  auto alloc_b = [&](size_t bytes) { char* r = wsp; wsp += (bytes + 255) & ~(size_t)255; return r; };
  auto af = [&](size_t n) { return (float*)alloc_b(n * sizeof(float)); };
  auto ah = [&](size_t n) { return (_Float16*)alloc_b(n * sizeof(_Float16)); };

  _Float16* A_psix1 = ah((size_t)16 * 32);
  _Float16* A_psix2 = ah((size_t)16 * 160);
  _Float16* A_enc1  = ah((size_t)32 * 288);
  _Float16* A_enc2  = ah((size_t)64 * 288);
  _Float16* A_encm  = ah((size_t)64 * 576);
  _Float16* A_encv  = ah((size_t)64 * 576);
  _Float16* A_pri   = ah((size_t)32 * 160);
  _Float16* A_prim  = ah((size_t)64 * 288);
  _Float16* A_priv  = ah((size_t)64 * 288);
  _Float16* A_psiz  = ah((size_t)16 * 576);
  _Float16* A_wi    = ah((size_t)3072 * 3072);
  _Float16* A_wh    = ah((size_t)3072 * 1024);
  _Float16* Bc      = ah((size_t)576 * 8192);   // largest im2col panel (encm/encv)
  _Float16* giT     = ah((size_t)B * 3072);
  _Float16* hT      = ah((size_t)B * 1024);

  float* px1   = af((size_t)B * 16 * 256);
  float* px2   = af((size_t)B * 16 * 64);
  float* e1    = af((size_t)B * 32 * 64);
  float* e2    = af((size_t)B * 64 * 64);
  float* zm    = af((size_t)B * 64 * 64);
  float* zlv   = af((size_t)B * 64 * 64);
  float* pm    = af((size_t)B * 64 * 64);
  float* plv   = af((size_t)B * 64 * 64);
  float* z     = af((size_t)B * 64 * 64);
  float* pbuf  = af((size_t)B * 32 * 64);
  float* zpsi  = af((size_t)B * 16 * 64);
  float* d1    = af((size_t)B * 32 * 225);
  float* decb[2] = { af((size_t)B * 3 * 1024), af((size_t)B * 3 * 1024) };
  float* hb[2]   = { af((size_t)B * 1024),     af((size_t)B * 1024) };
  float* gih   = af((size_t)3072 * B);
  float* ghh   = af((size_t)3072 * B);

  auto blk = [](long n) { return dim3((unsigned)((n + 255) / 256)); };

  auto wswz = [&](const float* src, int M, int K, int Kp, _Float16* dst) {
    wswz_k<<<blk((long)M * Kp), dim3(256), 0, stream>>>(src, M, K, Kp, dst);
  };
  auto gemm = [&](const _Float16* A, const _Float16* Bm, const float* bias, float* o,
                  int M, int N, int Kp, int act, int layout, int S) {
    int tiles = (M / 16) * (N / 64);
    wmma_gemm_k<<<dim3((tiles + GEMM_WAVES - 1) / GEMM_WAVES),
                  dim3(GEMM_WAVES * 32), 0, stream>>>(A, Bm, bias, o, M, N, Kp, act, layout, S);
  };
  auto im2col = [&](const float* s1, int C1, long bstr, long off, const float* s2, int C2,
                    int H, int W, int OH, int OW, int stride, int pad, int Kp) {
    im2col_k<<<blk((long)B * OH * OW * Kp), dim3(256), 0, stream>>>(
        s1, C1, bstr, off, s2, C2, H, W, OH, OW, stride, pad, B, Kp, Bc);
  };

  // ---- one-time weight conversion to row-major padded f16 A panels ----
  wswz(psix1_w, 16,   27, 32,  A_psix1);
  wswz(psix2_w, 16,  144, 160, A_psix2);
  wswz(enc1_w,  32,  288, 288, A_enc1);
  wswz(enc2_w,  64,  288, 288, A_enc2);
  wswz(encm_w,  64,  576, 576, A_encm);
  wswz(encv_w,  64,  576, 576, A_encv);
  wswz(pri_w,   32,  144, 160, A_pri);
  wswz(prim_w,  64,  288, 288, A_prim);
  wswz(priv_w,  64,  288, 288, A_priv);
  wswz(psiz_w,  16,  576, 576, A_psiz);
  wswz(gru_wi, 3072, 3072, 3072, A_wi);
  wswz(gru_wh, 3072, 1024, 1024, A_wh);

  zero_k<<<blk((long)B * 1024), dim3(256), 0, stream>>>(hb[0], (long)B * 1024);

  for (int t = 0; t < NSTEP; ++t) {
    float* hprev   = hb[t & 1];
    float* hnext   = hb[(t + 1) & 1];
    float* deccur  = decb[t & 1];
    float* decprev = decb[(t + 1) & 1];

    // psix1: 3ch 32x32 -> 16ch 16x16 (stride 2); teacher forcing for t<6
    if (t < 6) im2col(x, 3, (long)T * 3 * 1024, (long)t * 3 * 1024, nullptr, 0,
                      32, 32, 16, 16, 2, 1, 32);
    else       im2col(decprev, 3, (long)3 * 1024, 0, nullptr, 0,
                      32, 32, 16, 16, 2, 1, 32);
    gemm(A_psix1, Bc, psix1_b, px1, 16, B * 256, 32, ACT_RELU, 1, 256);

    // psix2: 16ch 16x16 -> 16ch 8x8 (stride 2)
    im2col(px1, 16, (long)16 * 256, 0, nullptr, 0, 16, 16, 8, 8, 2, 1, 160);
    gemm(A_psix2, Bc, psix2_b, px2, 16, B * 64, 160, ACT_RELU, 1, 64);

    // enc1: concat(px2, h) 32ch -> 32ch
    im2col(px2, 16, (long)16 * 64, 0, hprev, 16, 8, 8, 8, 8, 1, 1, 288);
    gemm(A_enc1, Bc, enc1_b, e1, 32, B * 64, 288, ACT_RELU, 1, 64);

    // enc2: 32ch -> 64ch
    im2col(e1, 32, (long)32 * 64, 0, nullptr, 0, 8, 8, 8, 8, 1, 1, 288);
    gemm(A_enc2, Bc, enc2_b, e2, 64, B * 64, 288, ACT_RELU, 1, 64);

    // encm / encv share im2col(e2)
    im2col(e2, 64, (long)64 * 64, 0, nullptr, 0, 8, 8, 8, 8, 1, 1, 576);
    gemm(A_encm, Bc, encm_b, zm,  64, B * 64, 576, ACT_NONE,     1, 64);
    gemm(A_encv, Bc, encv_b, zlv, 64, B * 64, 576, ACT_SOFTPLUS, 1, 64);

    // prior: h -> 32ch, then prim/priv share im2col(p)
    im2col(hprev, 16, (long)16 * 64, 0, nullptr, 0, 8, 8, 8, 8, 1, 1, 160);
    gemm(A_pri, Bc, pri_b, pbuf, 32, B * 64, 160, ACT_RELU, 1, 64);
    im2col(pbuf, 32, (long)32 * 64, 0, nullptr, 0, 8, 8, 8, 8, 1, 1, 288);
    gemm(A_prim, Bc, prim_b, pm,  64, B * 64, 288, ACT_NONE,     1, 64);
    gemm(A_priv, Bc, priv_b, plv, 64, B * 64, 288, ACT_SOFTPLUS, 1, 64);

    // z = zm + zlv * eps[t]
    zcomb_k<<<blk((long)B * 64 * 64), dim3(256), 0, stream>>>(
        zm, zlv, eps + (long)t * B * 64 * 64, z, (long)B * 64 * 64);

    // psiz: 64ch -> 16ch
    im2col(z, 64, (long)64 * 64, 0, nullptr, 0, 8, 8, 8, 8, 1, 1, 576);
    gemm(A_psiz, Bc, psiz_b, zpsi, 16, B * 64, 576, ACT_RELU, 1, 64);

    // decoder transposed convs (direct, LDS-staged weights, fused activations)
    convt1_k<<<blk((long)B * 32 * 225), dim3(256), 0, stream>>>(
        hprev, zpsi, dec1_w, dec1_b, d1, B);
    convt2_k<<<blk((long)B * 3 * 1024), dim3(256), 0, stream>>>(
        d1, dec2_w, dec2_b, deccur, out, B, t);

    // GRU: pack inputs (col-major f16), two WMMA GEMMs, then gate fusion
    pack_gi_k<<<blk((long)B * 3072), dim3(256), 0, stream>>>(px2, zpsi, hprev, giT, B);
    f32tof16_k<<<blk((long)B * 1024), dim3(256), 0, stream>>>(hprev, hT, (long)B * 1024);
    gemm(A_wi, giT, gru_bi, gih, 3072, B, 3072, ACT_NONE, 0, 0);
    gemm(A_wh, hT,  gru_bh, ghh, 3072, B, 1024, ACT_NONE, 0, 0);
    gruc_k<<<blk((long)B * 1024), dim3(256), 0, stream>>>(gih, ghh, hprev, hnext, B);
  }

  // tuple tail: zm, zlv, pm, plv of the final step
  const size_t OUT0 = (size_t)B * NSTEP * 3 * 1024;
  const size_t SZ   = (size_t)B * 64 * 64;
  hipMemcpyAsync(out + OUT0,          zm,  SZ * sizeof(float), hipMemcpyDeviceToDevice, stream);
  hipMemcpyAsync(out + OUT0 + SZ,     zlv, SZ * sizeof(float), hipMemcpyDeviceToDevice, stream);
  hipMemcpyAsync(out + OUT0 + 2 * SZ, pm,  SZ * sizeof(float), hipMemcpyDeviceToDevice, stream);
  hipMemcpyAsync(out + OUT0 + 3 * SZ, plv, SZ * sizeof(float), hipMemcpyDeviceToDevice, stream);
}
